// GPT_47966194762033
// MI455X (gfx1250) — matrix operand order repaired and checked by reading, not compiled, not run on hardware
//
#include <hip/hip_runtime.h>
#include <hip/hip_bf16.h>
#include <math.h>

// ---------------- model constants ----------------
static constexpr int LNUM = 8;
static constexpr int HN   = 16;
static constexpr int E    = 1024;
static constexpr int T    = 512;
static constexpr int BB   = 16;
static constexpr int HF   = 4;
static constexpr int S    = T + 1;      // 513
static constexpr int NT   = BB * S;     // 8208 tokens

typedef __bf16 v16bf __attribute__((ext_vector_type(16)));
typedef __bf16 v8bf  __attribute__((ext_vector_type(8)));
typedef __bf16 v4bf  __attribute__((ext_vector_type(4)));
typedef float  v8f   __attribute__((ext_vector_type(8)));

__device__ __forceinline__ v8f wmma_bf16(v16bf a, v16bf b, v8f c) {
    return __builtin_amdgcn_wmma_f32_16x16x32_bf16(
        false, a, false, b, (short)0, c, false, false);
}

__device__ __forceinline__ v8bf cvt8(float4 a, float4 b) {
    v8bf r;
    r[0] = (__bf16)a.x; r[1] = (__bf16)a.y; r[2] = (__bf16)a.z; r[3] = (__bf16)a.w;
    r[4] = (__bf16)b.x; r[5] = (__bf16)b.y; r[6] = (__bf16)b.z; r[7] = (__bf16)b.w;
    return r;
}
__device__ __forceinline__ v4bf cvt4(float4 a) {
    v4bf r;
    r[0] = (__bf16)a.x; r[1] = (__bf16)a.y; r[2] = (__bf16)a.z; r[3] = (__bf16)a.w;
    return r;
}
__device__ __forceinline__ v16bf cat16(v8bf lo, v8bf hi) {
    return __builtin_shufflevector(lo, hi, 0,1,2,3,4,5,6,7,8,9,10,11,12,13,14,15);
}

// ---------------- embedding ----------------
__global__ __launch_bounds__(256) void embed_kernel(
    const int* __restrict__ idx, const float* __restrict__ xT,
    const float* __restrict__ tok_emb, const float* __restrict__ pos_emb,
    const float* __restrict__ xT_w, const float* __restrict__ xT_b,
    float* __restrict__ x)
{
    long gid = (long)blockIdx.x * 256 + threadIdx.x;
    long total = (long)NT * E;
    if (gid >= total) return;
    int e  = (int)(gid % E);
    long tk = gid / E;
    int b = (int)(tk / S);
    int s = (int)(tk % S);
    float val;
    if (s < T) {
        int tok = idx[b * T + s];
        val = tok_emb[(long)tok * E + e];
    } else {
        val = xT[b * 2 + 0] * xT_w[0 * E + e] + xT[b * 2 + 1] * xT_w[1 * E + e] + xT_b[e];
    }
    x[gid] = val + pos_emb[(long)s * E + e];
}

// ---------------- layernorm (one token per block) ----------------
__global__ __launch_bounds__(256) void ln_kernel(
    const float* __restrict__ x, const float* __restrict__ g,
    const float* __restrict__ bta, float* __restrict__ out)
{
    __shared__ float s1[256], s2[256];
    int t = threadIdx.x;
    long base = (long)blockIdx.x * E;
    float a0 = x[base + t], a1 = x[base + t + 256];
    float a2 = x[base + t + 512], a3 = x[base + t + 768];
    s1[t] = a0 + a1 + a2 + a3;
    s2[t] = a0 * a0 + a1 * a1 + a2 * a2 + a3 * a3;
    __syncthreads();
    for (int off = 128; off > 0; off >>= 1) {
        if (t < off) { s1[t] += s1[t + off]; s2[t] += s2[t + off]; }
        __syncthreads();
    }
    float mean = s1[0] * (1.0f / E);
    float var  = s2[0] * (1.0f / E) - mean * mean;
    float inv  = rsqrtf(var + 1e-5f);
    out[base + t]       = (a0 - mean) * inv * g[t]       + bta[t];
    out[base + t + 256] = (a1 - mean) * inv * g[t + 256] + bta[t + 256];
    out[base + t + 512] = (a2 - mean) * inv * g[t + 512] + bta[t + 512];
    out[base + t + 768] = (a3 - mean) * inv * g[t + 768] + bta[t + 768];
}

// ---------------- weight pack: fp32 [K][N] -> bf16 [N][K] (32x32 LDS transpose) ----------------
__global__ __launch_bounds__(256) void wconv_kernel(
    const float* __restrict__ W, __bf16* __restrict__ Wt, int K, int N)
{
    __shared__ float tile[32][33];
    int tx = threadIdx.x & 31, ty = threadIdx.x >> 5;   // ty 0..7
    int n0 = blockIdx.x * 32, k0 = blockIdx.y * 32;
    #pragma unroll
    for (int j = 0; j < 4; j++)
        tile[ty + j * 8][tx] = W[(long)(k0 + ty + j * 8) * N + n0 + tx];
    __syncthreads();
    #pragma unroll
    for (int j = 0; j < 4; j++)
        Wt[(long)(n0 + ty + j * 8) * K + k0 + tx] = (__bf16)tile[tx][ty + j * 8];
}

// ---------------- WMMA GEMM: C = act(A[M,K] @ W[K,N] + bias) (+ res) ----------------
// W pre-packed bf16, transposed [N][K]. Block tile 128x128, 8 waves (4Mx2N),
// each wave 32x64 = 2x4 WMMA tiles. Double-buffered LDS; fragments are
// contiguous 16B runs -> ds_load_b128 pairs. A rows clamped (no staging guards).
// act: 0 = none, 1 = exact GELU, 2 = ReLU
__global__ __launch_bounds__(256) void gemm_kernel(
    const float* __restrict__ A, const __bf16* __restrict__ Wt,
    const float* __restrict__ bias, const float* __restrict__ res,
    float* __restrict__ C, int M, int N, int K, int act)
{
    constexpr int BM = 128, BN = 128, BK = 32, LDP = 40;
    __shared__ __bf16 As[2][BM][LDP];
    __shared__ __bf16 Bt[2][BN][LDP];

    int tid  = threadIdx.x;
    int lane = tid & 31;
    int wave = tid >> 5;
    int wm   = (wave & 3) * 32;
    int wn   = (wave >> 2) * 64;
    int hl   = lane >> 4;
    int l16  = lane & 15;
    int bm0  = blockIdx.x * BM;
    int bn0  = blockIdx.y * BN;

    // staging geometry (k0-invariant): A = 4 float4 chunks, B = 2 b128 copies
    int  asr[4], acol[4];
    long arow[4];
    #pragma unroll
    for (int i = 0; i < 4; i++) {
        int ci = tid + i * 256;           // 1024 chunks of 4 over 128x32
        asr[i]  = ci >> 3;
        acol[i] = (ci & 7) * 4;
        int gr = bm0 + asr[i];
        if (gr > M - 1) gr = M - 1;       // clamp: junk rows never stored
        arow[i] = (long)gr * K;
    }
    int  sn   = tid & 127;
    int  sk   = (tid >> 7) * 16;
    long brow = (long)(bn0 + sn) * K;

    float4 fa[4];
    v8bf   wb0, wb1;
    auto load_regs = [&](int k0) {
        #pragma unroll
        for (int i = 0; i < 4; i++)
            fa[i] = *(const float4*)&A[arow[i] + k0 + acol[i]];
        wb0 = *(const v8bf*)&Wt[brow + k0 + sk];
        wb1 = *(const v8bf*)&Wt[brow + k0 + sk + 8];
    };
    auto store_regs = [&](int buf) {
        #pragma unroll
        for (int i = 0; i < 4; i++)
            *(v4bf*)&As[buf][asr[i]][acol[i]] = cvt4(fa[i]);
        *(v8bf*)&Bt[buf][sn][sk]     = wb0;
        *(v8bf*)&Bt[buf][sn][sk + 8] = wb1;
    };

    v8f acc[8] = {};
    int nk = K / BK;

    load_regs(0);
    store_regs(0);
    __syncthreads();

    for (int kc = 0; kc < nk; kc++) {
        int cur = kc & 1;
        if (kc + 1 < nk) load_regs((kc + 1) * BK);   // hide global latency

        v16bf af[2], bfr[4];
        #pragma unroll
        for (int im = 0; im < 2; im++) {
            int row = wm + im * 16 + l16;
            af[im] = cat16(*(const v8bf*)&As[cur][row][hl * 8],
                           *(const v8bf*)&As[cur][row][16 + hl * 8]);
        }
        #pragma unroll
        for (int jn = 0; jn < 4; jn++) {
            int col = wn + jn * 16 + l16;
            bfr[jn] = cat16(*(const v8bf*)&Bt[cur][col][hl * 8],
                            *(const v8bf*)&Bt[cur][col][16 + hl * 8]);
        }
        #pragma unroll
        for (int im = 0; im < 2; im++)
            #pragma unroll
            for (int jn = 0; jn < 4; jn++)
                acc[im * 4 + jn] = wmma_bf16(af[im], bfr[jn], acc[im * 4 + jn]);

        if (kc + 1 < nk) store_regs(cur ^ 1);
        __syncthreads();
    }

    // ---- epilogue ----
    bool fullM = (bm0 + BM) <= M;
    #pragma unroll
    for (int im = 0; im < 2; im++)
        #pragma unroll
        for (int jn = 0; jn < 4; jn++) {
            v8f cc = acc[im * 4 + jn];
            int n  = bn0 + wn + jn * 16 + l16;
            float bv = bias[n];
            #pragma unroll
            for (int r = 0; r < 8; r++) {
                int m = bm0 + wm + im * 16 + hl * 8 + r;
                if (fullM || m < M) {
                    float v = cc[r] + bv;
                    if (act == 1)      v = 0.5f * v * (1.f + erff(v * 0.70710678118f));
                    else if (act == 2) v = fmaxf(v, 0.f);
                    if (res) v += res[(long)m * N + n];
                    C[(long)m * N + n] = v;
                }
            }
        }
}

// ---------------- attention: one wave per (b, h, 16-query tile) ----------------
__global__ __launch_bounds__(32) void attn_kernel(
    const float* __restrict__ q, const float* __restrict__ k,
    const float* __restrict__ v, float* __restrict__ y)
{
    constexpr int SP = 544;                 // S padded to 34*16
    __shared__ float sc[16][SP];
    int lane = threadIdx.x;
    int hl = lane >> 4, l16 = lane & 15;
    int qt = blockIdx.x;                    // 0..32
    int h  = blockIdx.y;
    int b  = blockIdx.z;
    const float scale = 0.125f;             // 1/sqrt(64)

    // q fragments: one contiguous 16-float run per frag -> float4 loads
    v16bf qa[2];
    int qrow = qt * 16 + l16; if (qrow >= S) qrow = S - 1;
    long qbase = ((long)(b * S + qrow)) * E + h * 64;
    #pragma unroll
    for (int j = 0; j < 2; j++) {
        const float* p = &q[qbase + j * 32];
        float4 lo0 = *(const float4*)&p[hl * 16];
        float4 lo1 = *(const float4*)&p[hl * 16 + 4];
        float4 hi0 = *(const float4*)&p[hl * 16 + 8];
        float4 hi1 = *(const float4*)&p[hl * 16 + 12];
        qa[j] = cat16(cvt8(lo0, lo1), cvt8(hi0, hi1));
    }

    // phase 1: scores[16, S] -> LDS. Pad-column scores (key>=S) are garbage but
    // never read: softmax scans c<S and zeroes the pad afterwards.
    for (int kc = 0; kc < 34; kc++) {
        int key = kc * 16 + l16;
        int keyc = (key < S) ? key : (S - 1);       // clamp for address safety
        const float* kp = &k[((long)(b * S + keyc)) * E + h * 64];
        v16bf b0, b1;
        {
            float4 a0 = *(const float4*)&kp[hl * 16];
            float4 a1 = *(const float4*)&kp[hl * 16 + 4];
            float4 a2 = *(const float4*)&kp[hl * 16 + 8];
            float4 a3 = *(const float4*)&kp[hl * 16 + 12];
            b0 = cat16(cvt8(a0, a1), cvt8(a2, a3));
            float4 c0 = *(const float4*)&kp[32 + hl * 16];
            float4 c1 = *(const float4*)&kp[32 + hl * 16 + 4];
            float4 c2 = *(const float4*)&kp[32 + hl * 16 + 8];
            float4 c3 = *(const float4*)&kp[32 + hl * 16 + 12];
            b1 = cat16(cvt8(c0, c1), cvt8(c2, c3));
        }
        v8f s = {};
        s = wmma_bf16(qa[0], b0, s);
        s = wmma_bf16(qa[1], b1, s);
        #pragma unroll
        for (int r = 0; r < 8; r++)
            sc[hl * 8 + r][kc * 16 + l16] = s[r] * scale;
    }
    __syncthreads();

    // phase 2: row softmax (lanes 0..15, one row each); zero the pad columns
    if (lane < 16) {
        float mx = -1e30f;
        for (int c = 0; c < S; c++) mx = fmaxf(mx, sc[lane][c]);
        float sum = 0.f;
        for (int c = 0; c < S; c++) { float e = __expf(sc[lane][c] - mx); sc[lane][c] = e; sum += e; }
        float inv = 1.f / sum;
        for (int c = 0; c < S; c++)  sc[lane][c] *= inv;
        for (int c = S; c < SP; c++) sc[lane][c] = 0.f;
    }
    __syncthreads();

    // phase 3: y = P @ V, 17 chunks of K=32, 4 output tiles of 16 cols
    v8f acc[4] = {};
    for (int kc = 0; kc < 17; kc++) {
        v16bf pa;
        {
            const float* pr = &sc[l16][kc * 32];
            float4 lo0 = *(const float4*)&pr[hl * 8];
            float4 lo1 = *(const float4*)&pr[hl * 8 + 4];
            float4 hi0 = *(const float4*)&pr[16 + hl * 8];
            float4 hi1 = *(const float4*)&pr[16 + hl * 8 + 4];
            pa = cat16(cvt8(lo0, lo1), cvt8(hi0, hi1));
        }
        #pragma unroll
        for (int jn = 0; jn < 4; jn++) {
            v16bf vb;
            #pragma unroll
            for (int i = 0; i < 16; i++) {
                int key = kc * 32 + hl * 8 + (i & 7) + (i >> 3) * 16;
                int kcl = (key < S) ? key : (S - 1);   // P==0 on pad rows
                vb[i] = (__bf16)v[((long)(b * S + kcl)) * E + h * 64 + jn * 16 + l16];
            }
            acc[jn] = wmma_bf16(pa, vb, acc[jn]);
        }
    }
    #pragma unroll
    for (int jn = 0; jn < 4; jn++)
        #pragma unroll
        for (int r = 0; r < 8; r++) {
            int m = qt * 16 + hl * 8 + r;
            if (m < S)
                y[((long)(b * S + m)) * E + h * 64 + jn * 16 + l16] = acc[jn][r];
        }
}

// ---------------- token max-pool over S ----------------
__global__ __launch_bounds__(256) void maxpool_kernel(
    const float* __restrict__ hbuf, float* __restrict__ xm)
{
    int gid = blockIdx.x * 256 + threadIdx.x;
    if (gid >= BB * E) return;
    int b = gid / E, e = gid % E;
    float m = -1e30f;
    for (int s = 0; s < S; s++)
        m = fmaxf(m, hbuf[((long)(b * S + s)) * E + e]);
    xm[gid] = m;
}

// ---------------- head: logits[b] = xr[b,:] . head_w + head_b ----------------
__global__ __launch_bounds__(256) void head_kernel(
    const float* __restrict__ xr, const float* __restrict__ hw,
    const float* __restrict__ hb, float* __restrict__ out)
{
    __shared__ float red[256];
    int b = blockIdx.x, t = threadIdx.x;
    float s = 0.f;
    for (int e = t; e < E; e += 256) s += xr[b * E + e] * hw[e];
    red[t] = s; __syncthreads();
    for (int off = 128; off > 0; off >>= 1) {
        if (t < off) red[t] += red[t + off];
        __syncthreads();
    }
    if (t == 0) out[b] = red[0] + hb[0];
}

// ---------------- host launcher ----------------
extern "C" void kernel_launch(void* const* d_in, const int* in_sizes, int n_in,
                              void* d_out, int out_size, void* d_ws, size_t ws_size,
                              hipStream_t stream)
{
    const int*   idx     = (const int*)  d_in[0];
    const float* xT      = (const float*)d_in[1];
    const float* tok_emb = (const float*)d_in[2];
    const float* pos_emb = (const float*)d_in[3];
    const float* xT_w    = (const float*)d_in[4];
    const float* xT_b    = (const float*)d_in[5];
    const float* ln1_g   = (const float*)d_in[6];
    const float* ln1_b   = (const float*)d_in[7];
    const float* Wq      = (const float*)d_in[8];
    const float* bq      = (const float*)d_in[9];
    const float* Wk      = (const float*)d_in[10];
    const float* bk      = (const float*)d_in[11];
    const float* Wv      = (const float*)d_in[12];
    const float* bv      = (const float*)d_in[13];
    const float* Wp      = (const float*)d_in[14];
    const float* bp      = (const float*)d_in[15];
    const float* ln2_g   = (const float*)d_in[16];
    const float* ln2_b   = (const float*)d_in[17];
    const float* Wm1     = (const float*)d_in[18];
    const float* bm1     = (const float*)d_in[19];
    const float* Wm2     = (const float*)d_in[20];
    const float* bm2     = (const float*)d_in[21];
    const float* lnf_g   = (const float*)d_in[22];
    const float* lnf_b   = (const float*)d_in[23];
    const float* dec_w   = (const float*)d_in[24];
    const float* dec_b   = (const float*)d_in[25];
    const float* head_w  = (const float*)d_in[26];
    const float* head_b  = (const float*)d_in[27];
    (void)in_sizes; (void)n_in; (void)out_size; (void)ws_size;

    char* wp = (char*)d_ws;
    auto carve = [&](size_t bytes) -> void* {
        void* p = (void*)wp;
        wp += (bytes + 255) & ~(size_t)255;
        return p;
    };
    const size_t actB = (size_t)NT * E * sizeof(float);
    float*  x    = (float*)carve(actB);
    float*  hbuf = (float*)carve(actB);
    float*  qb   = (float*)carve(actB);
    float*  kb   = (float*)carve(actB);
    float*  vb   = (float*)carve(actB);
    float*  hid  = (float*)carve((size_t)NT * HF * E * sizeof(float));
    float*  xm   = (float*)carve((size_t)BB * E * sizeof(float));
    float*  xr   = (float*)carve((size_t)BB * E * sizeof(float));
    __bf16* wbuf = (__bf16*)carve((size_t)E * HF * E * sizeof(__bf16)); // largest weight

    long tot = (long)NT * E;
    embed_kernel<<<(unsigned)((tot + 255) / 256), 256, 0, stream>>>(
        idx, xT, tok_emb, pos_emb, xT_w, xT_b, x);

    dim3 gEE((NT + 127) / 128, E / 128);
    dim3 gEH((NT + 127) / 128, (HF * E) / 128);
    dim3 cEE(E / 32, E / 32);              // wconv grid for [E][E]
    dim3 cEH(HF * E / 32, E / 32);         // [E][4E] -> (N/32, K/32)
    dim3 cHE(E / 32, HF * E / 32);         // [4E][E]

    for (int l = 0; l < LNUM; l++) {
        const float* WqL  = Wq  + (size_t)l * E * E;
        const float* WkL  = Wk  + (size_t)l * E * E;
        const float* WvL  = Wv  + (size_t)l * E * E;
        const float* WpL  = Wp  + (size_t)l * E * E;
        const float* Wm1L = Wm1 + (size_t)l * E * HF * E;
        const float* Wm2L = Wm2 + (size_t)l * HF * E * E;

        ln_kernel<<<NT, 256, 0, stream>>>(x, ln1_g + l * E, ln1_b + l * E, hbuf);
        wconv_kernel<<<cEE, 256, 0, stream>>>(WqL, wbuf, E, E);
        gemm_kernel<<<gEE, 256, 0, stream>>>(hbuf, wbuf, bq + l * E, nullptr, qb, NT, E, E, 0);
        wconv_kernel<<<cEE, 256, 0, stream>>>(WkL, wbuf, E, E);
        gemm_kernel<<<gEE, 256, 0, stream>>>(hbuf, wbuf, bk + l * E, nullptr, kb, NT, E, E, 0);
        wconv_kernel<<<cEE, 256, 0, stream>>>(WvL, wbuf, E, E);
        gemm_kernel<<<gEE, 256, 0, stream>>>(hbuf, wbuf, bv + l * E, nullptr, vb, NT, E, E, 0);
        attn_kernel<<<dim3(33, HN, BB), 32, 0, stream>>>(qb, kb, vb, hbuf);
        wconv_kernel<<<cEE, 256, 0, stream>>>(WpL, wbuf, E, E);
        gemm_kernel<<<gEE, 256, 0, stream>>>(hbuf, wbuf, bp + l * E, x, x, NT, E, E, 0);
        ln_kernel<<<NT, 256, 0, stream>>>(x, ln2_g + l * E, ln2_b + l * E, qb);
        wconv_kernel<<<cEH, 256, 0, stream>>>(Wm1L, wbuf, E, HF * E);
        gemm_kernel<<<gEH, 256, 0, stream>>>(qb, wbuf, bm1 + (size_t)l * HF * E, nullptr,
                                             hid, NT, HF * E, E, 1);
        wconv_kernel<<<cHE, 256, 0, stream>>>(Wm2L, wbuf, HF * E, E);
        gemm_kernel<<<gEE, 256, 0, stream>>>(hid, wbuf, bm2 + l * E, x, x, NT, E, HF * E, 0);
    }

    ln_kernel<<<NT, 256, 0, stream>>>(x, lnf_g, lnf_b, hbuf);
    maxpool_kernel<<<(BB * E + 255) / 256, 256, 0, stream>>>(hbuf, xm);
    wconv_kernel<<<cEE, 256, 0, stream>>>(dec_w, wbuf, E, E);
    gemm_kernel<<<dim3(1, E / 128), 256, 0, stream>>>(xm, wbuf, dec_b, nullptr, xr, BB, E, E, 2);
    head_kernel<<<BB, 256, 0, stream>>>(xr, head_w, head_b, (float*)d_out);
}
